// RNN_81518479278207
// MI455X (gfx1250) — matrix operand order repaired and checked by reading, not compiled, not run on hardware
//
#include <hip/hip_runtime.h>
#include <hip/hip_bf16.h>
#include <math.h>

#define B_DIM 128
#define S_DIM 1024
#define EMB_DIM 128
#define HID_DIM 128
#define NLAB 9
#define VOCAB 32000

typedef __attribute__((ext_vector_type(16))) __bf16 v16bf;
typedef __attribute__((ext_vector_type(8)))  float  v8f;

// A frag 16x32 in this code's contiguous k order (B packed alike): lanes 0-15 -> K 0..15,
// lanes 16-31 -> K 16..31; row = lane & 15. One contiguous 32B run
// per lane -> compiler merges into 2x ds_load_b128.
__device__ __forceinline__ v16bf a_frag_from_lds(const __bf16* sA, int kChunk, int lane) {
  v16bf a;
  const int row = lane & 15;
  const int kb  = kChunk * 32 + ((lane & 16) ? 16 : 0);
#pragma unroll
  for (int i = 0; i < 16; ++i) {
    int k = kb + i;
    a[i] = sA[row * HID_DIM + k];
  }
  return a;
}

// ---- Kernel 0a: pack weights into WMMA B-fragment order (bf16) ----
// Packed layout: frag[kChunk][colTile][lane][i] contiguous, where the value is
// W[(kChunk*32 + (lane&16 ? 16:0) + i) * ldn + colTile*16 + (lane&15)].
__global__ void __launch_bounds__(256) rnn_prep_weights_kernel(
    const float* __restrict__ W_h, const float* __restrict__ W_out,
    __bf16* __restrict__ wxpk, __bf16* __restrict__ whpk,
    __bf16* __restrict__ wopk, float* __restrict__ lossPart) {
  int i = blockIdx.x * 256 + threadIdx.x;
  if (i < 16384) {                                   // Wx = W_h[:128] (128x128)
    int e = i & 15, lane = (i >> 4) & 31, ct = (i >> 9) & 7, kk = i >> 12;
    int k = kk * 32 + ((lane & 16) ? 16 : 0) + e;
    int col = ct * 16 + (lane & 15);
    wxpk[i] = (__bf16)W_h[k * HID_DIM + col];
  } else if (i < 32768) {                            // Wh = W_h[128:] (128x128)
    int o = i - 16384;
    int e = o & 15, lane = (o >> 4) & 31, ct = (o >> 9) & 7, kk = o >> 12;
    int k = kk * 32 + ((lane & 16) ? 16 : 0) + e;
    int col = ct * 16 + (lane & 15);
    whpk[o] = (__bf16)W_h[(HID_DIM + k) * HID_DIM + col];
  } else if (i < 34816) {                            // W_out 128x9 -> padded 128x16
    int o = i - 32768;
    int e = o & 15, lane = (o >> 4) & 31, kk = o >> 9;
    int k = kk * 32 + ((lane & 16) ? 16 : 0) + e;
    int col = lane & 15;
    wopk[o] = (__bf16)((col < NLAB) ? W_out[k * NLAB + col] : 0.0f);
  } else if (i < 34824) {
    lossPart[i - 34816] = 0.0f;
  }
}

// ---- Kernel 0b: emb_table f32 -> bf16 (16MB -> 8MB, L2-resident) ----
__global__ void __launch_bounds__(256) rnn_prep_emb_kernel(
    const float* __restrict__ emb, __bf16* __restrict__ embbf) {
  int i = blockIdx.x * 256 + threadIdx.x;
  if (i < VOCAB * EMB_DIM) embbf[i] = (__bf16)emb[i];
}

// ---- Kernel 1: xW[s][b][h] = embbf[ids[b,s]] @ Wx + b_h (time-major out) ----
__global__ void __launch_bounds__(256) rnn_embed_xw_kernel(
    const int* __restrict__ ids, const __bf16* __restrict__ embbf,
    const __bf16* __restrict__ wxpk, const float* __restrict__ b_h,
    float* __restrict__ xw_tm) {
  __shared__ __align__(16) __bf16 sA[16 * EMB_DIM];
  __shared__ int sIds[16];
  const int tid = threadIdx.x;
  const int g0  = blockIdx.x * 16;                   // 16 consecutive tokens, same b
  if (tid < 16) sIds[tid] = ids[g0 + tid];
  __syncthreads();
  {                                                  // b128 gather: 16B per thread
    int row = tid >> 4, seg = tid & 15;
    const uint4* src = (const uint4*)(embbf + (size_t)sIds[row] * EMB_DIM);
    ((uint4*)sA)[row * 16 + seg] = src[seg];
  }
  __syncthreads();

  const int wave = tid >> 5, lane = tid & 31;
  const int col = wave * 16 + (lane & 15);
  const v16bf* wxv = (const v16bf*)wxpk;
  v8f c = {};
#pragma unroll
  for (int kk = 0; kk < 4; ++kk) {
    v16bf a = a_frag_from_lds(sA, kk, lane);
    v16bf b = wxv[(kk * 8 + wave) * 32 + lane];      // contiguous 32B per lane
    c = __builtin_amdgcn_wmma_f32_16x16x32_bf16(false, a, false, b, (short)0, c, false, false);
  }
  const int bidx = g0 / S_DIM;
  const int s0   = g0 % S_DIM;
  const int hi   = (lane & 16) ? 8 : 0;
  const float bias = b_h[col];
#pragma unroll
  for (int j = 0; j < 8; ++j) {
    int s = s0 + j + hi;
    xw_tm[((size_t)s * B_DIM + bidx) * HID_DIM + col] = c[j] + bias;
  }
}

// ---- Kernel 2: sequential scan, 8 blocks x 16 batch rows, 8 waves per block.
// Software-pipelined: wave 0 computes logits/softmax for token t-1 (same sH
// everyone reads) overlapped with the h-update WMMAs. 2 barriers per step. ----
__global__ void __launch_bounds__(256) rnn_scan_kernel(
    const float* __restrict__ xw_tm,
    const __bf16* __restrict__ whpk,
    const __bf16* __restrict__ wopk,
    const float* __restrict__ b_out,
    const int* __restrict__ labels,
    float* __restrict__ logits_out,
    float* __restrict__ lossPart) {
  __shared__ __align__(16) __bf16 sH[16 * HID_DIM];
  __shared__ float sLog[16 * 17];                    // stride 17: conflict-free
  __shared__ float sRed[16];
  const int tid = threadIdx.x, wave = tid >> 5, lane = tid & 31;
  const int rb  = blockIdx.x * 16;                   // batch rows [rb, rb+16)
  const int col = wave * 16 + (lane & 15);
  const int hi  = (lane & 16) ? 8 : 0;

  const v16bf* whv = (const v16bf*)whpk;
  const v16bf* wov = (const v16bf*)wopk;
  v16bf bw[4];                                       // loop-invariant Wh frags
#pragma unroll
  for (int kk = 0; kk < 4; ++kk) bw[kk] = whv[(kk * 8 + wave) * 32 + lane];
  v16bf bo[4];                                       // W_out frags (wave 0)
  if (wave == 0) {
#pragma unroll
    for (int kk = 0; kk < 4; ++kk) bo[kk] = wov[kk * 32 + lane];
  }

  for (int p = tid; p < 16 * HID_DIM; p += 256) sH[p] = (__bf16)0.0f;  // h_{-1}=0
  float lossAcc = 0.0f;
  __syncthreads();

  for (int t = 0; t < S_DIM; ++t) {
    // h_{t-1} @ Wh (all 8 waves)
    v8f c = {};
#pragma unroll
    for (int kk = 0; kk < 4; ++kk) {
      v16bf a = a_frag_from_lds(sH, kk, lane);
      c = __builtin_amdgcn_wmma_f32_16x16x32_bf16(false, a, false, bw[kk], (short)0, c, false, false);
    }
    float xv[8];
#pragma unroll
    for (int j = 0; j < 8; ++j) {                    // coalesced time-major read
      const float* p = &xw_tm[((size_t)t * B_DIM + (rb + j + hi)) * HID_DIM + col];
      xv[j] = *p;
      __builtin_prefetch(p + (size_t)B_DIM * HID_DIM, 0, 3);  // next step's tile
    }

    if (wave == 0 && t > 0) {                        // logits(t-1) = h_{t-1} @ Wout
      v8f cl = {};
#pragma unroll
      for (int kk = 0; kk < 4; ++kk) {
        v16bf a = a_frag_from_lds(sH, kk, lane);
        cl = __builtin_amdgcn_wmma_f32_16x16x32_bf16(false, a, false, bo[kk], (short)0, cl, false, false);
      }
#pragma unroll
      for (int j = 0; j < 8; ++j) sLog[(j + hi) * 17 + (lane & 15)] = cl[j];
      if (lane < 16) {                               // same wave: LDS in-order
        float v[NLAB];
        float m = -1e30f;
#pragma unroll
        for (int cc = 0; cc < NLAB; ++cc) {
          v[cc] = sLog[lane * 17 + cc] + b_out[cc];
          m = fmaxf(m, v[cc]);
        }
        float sum = 0.0f;
#pragma unroll
        for (int cc = 0; cc < NLAB; ++cc) sum += __expf(v[cc] - m);
        float lse = __logf(sum) + m;
        size_t token = (size_t)(rb + lane) * S_DIM + (t - 1);
#pragma unroll
        for (int cc = 0; cc < NLAB; ++cc) logits_out[token * NLAB + cc] = v[cc];
        lossAcc += (lse - v[labels[token]]);
      }
    }
    __syncthreads();                                 // B1: all done reading h_{t-1}
#pragma unroll
    for (int j = 0; j < 8; ++j)
      sH[(j + hi) * HID_DIM + col] = (__bf16)tanhf(xv[j] + c[j]);
    __syncthreads();                                 // B2: sH = h_t
  }

  if (wave == 0) {                                   // epilogue: token S-1
    v8f cl = {};
#pragma unroll
    for (int kk = 0; kk < 4; ++kk) {
      v16bf a = a_frag_from_lds(sH, kk, lane);
      cl = __builtin_amdgcn_wmma_f32_16x16x32_bf16(false, a, false, bo[kk], (short)0, cl, false, false);
    }
#pragma unroll
    for (int j = 0; j < 8; ++j) sLog[(j + hi) * 17 + (lane & 15)] = cl[j];
    if (lane < 16) {
      float v[NLAB];
      float m = -1e30f;
#pragma unroll
      for (int cc = 0; cc < NLAB; ++cc) {
        v[cc] = sLog[lane * 17 + cc] + b_out[cc];
        m = fmaxf(m, v[cc]);
      }
      float sum = 0.0f;
#pragma unroll
      for (int cc = 0; cc < NLAB; ++cc) sum += __expf(v[cc] - m);
      float lse = __logf(sum) + m;
      size_t token = (size_t)(rb + lane) * S_DIM + (S_DIM - 1);
#pragma unroll
      for (int cc = 0; cc < NLAB; ++cc) logits_out[token * NLAB + cc] = v[cc];
      lossAcc += (lse - v[labels[token]]);
    }
  }

  if (tid < 16) sRed[tid] = lossAcc;
  __syncthreads();
  if (tid == 0) {
    float s = 0.0f;
    for (int i = 0; i < 16; ++i) s += sRed[i];
    lossPart[blockIdx.x] = s;
  }
}

// ---- Kernel 3: reduce 8 partial losses -> mean NLL ----
__global__ void rnn_finalize_kernel(const float* __restrict__ lossPart,
                                    float* __restrict__ out_loss) {
  float s = 0.0f;
  for (int i = 0; i < B_DIM / 16; ++i) s += lossPart[i];
  *out_loss = s / (float)((size_t)B_DIM * S_DIM);
}

extern "C" void kernel_launch(void* const* d_in, const int* in_sizes, int n_in,
                              void* d_out, int out_size, void* d_ws, size_t ws_size,
                              hipStream_t stream) {
  const int*   input_ids = (const int*)d_in[0];
  const int*   labels    = (const int*)d_in[3];
  const float* emb_table = (const float*)d_in[4];
  const float* W_h       = (const float*)d_in[5];
  const float* b_h       = (const float*)d_in[6];
  const float* W_out     = (const float*)d_in[7];
  const float* b_out     = (const float*)d_in[8];
  float* out = (float*)d_out;

  char* ws = (char*)d_ws;
  __bf16* wxpk     = (__bf16*)(ws + 0);              // packed Wx frags (32 KB)
  __bf16* whpk     = (__bf16*)(ws + 32768);          // packed Wh frags (32 KB)
  __bf16* wopk     = (__bf16*)(ws + 65536);          // packed Wout frags (4 KB)
  float*  lossPart = (float*)(ws + 69632);           // 8 floats
  __bf16* embbf    = (__bf16*)(ws + 131072);         // 32000x128 bf16 (~7.8 MB)
  float*  xw_tm    = (float*)(ws + (size_t)8388608); // [S][B][HID] f32 (64 MB)

  rnn_prep_weights_kernel<<<137, 256, 0, stream>>>(W_h, W_out, wxpk, whpk, wopk, lossPart);
  rnn_prep_emb_kernel<<<(VOCAB * EMB_DIM) / 256, 256, 0, stream>>>(emb_table, embbf);
  rnn_embed_xw_kernel<<<(B_DIM * S_DIM) / 16, 256, 0, stream>>>(
      input_ids, embbf, wxpk, b_h, xw_tm);
  rnn_scan_kernel<<<B_DIM / 16, 256, 0, stream>>>(
      xw_tm, whpk, wopk, b_out, labels, out, lossPart);
  rnn_finalize_kernel<<<1, 1, 0, stream>>>(
      lossPart, out + (size_t)B_DIM * S_DIM * NLAB);
}
